// ComplexSelfAttention_3298534884199
// MI455X (gfx1250) — matrix operand order repaired and checked by reading, not compiled, not run on hardware
//
#include <hip/hip_runtime.h>
#include <stdint.h>

#define M_ROWS 4096   // B*N
#define C_DIM  1024
#define K_DIM  2048   // 2*C  (packed [re|im])
#define HEADS  16
#define HDIM   64
#define SEQ    2048
#define BATCH  2

typedef __attribute__((ext_vector_type(16))) __bf16       bf16x16;
typedef __attribute__((ext_vector_type(8)))  float        f32x8;
typedef __attribute__((ext_vector_type(4)))  unsigned int u32x4;

union Frag32B { u32x4 q[2]; bf16x16 v; };        // one WMMA A/B operand (32 bytes/lane)
union AccU    { f32x8 v; float f[8]; };          // one WMMA C/D accumulator
union Pack16  { u32x4 q[2]; unsigned short s[16]; };

__device__ __forceinline__ unsigned short f2bf(float x) {
  unsigned int u = __float_as_uint(x);
  unsigned int r = 0x7FFFu + ((u >> 16) & 1u);   // round-to-nearest-even
  return (unsigned short)((u + r) >> 16);
}

// ---------------- packing kernels (fp32 -> bf16, complex-linear folding) ------------

__global__ void pack_x_kernel(const float* __restrict__ xre,
                              const float* __restrict__ xim,
                              unsigned short* __restrict__ X) {
  size_t i = (size_t)blockIdx.x * 256 + threadIdx.x;   // over M_ROWS*K_DIM
  size_t row = i >> 11;
  size_t k   = i & 2047;
  float v = (k < C_DIM) ? xre[row * C_DIM + k] : xim[row * C_DIM + (k - C_DIM)];
  X[i] = f2bf(v);
}

__global__ void pack_w_kernel(const float* __restrict__ Wre, const float* __restrict__ bre,
                              const float* __restrict__ Wim, const float* __restrict__ bim,
                              unsigned short* __restrict__ WcRe, unsigned short* __restrict__ WcIm,
                              float* __restrict__ biasRe, float* __restrict__ biasIm) {
  size_t i = (size_t)blockIdx.x * 256 + threadIdx.x;   // over C_DIM*K_DIM
  size_t n = i >> 11;
  size_t k = i & 2047;
  if (k < C_DIM) {
    float wre = Wre[n * C_DIM + k];
    float wim = Wim[n * C_DIM + k];
    WcRe[i] = f2bf(wre);          // re path: [Wre | -Wim]
    WcIm[i] = f2bf(wim);          // im path: [Wim |  Wre]
  } else {
    size_t kk = k - C_DIM;
    float wre = Wre[n * C_DIM + kk];
    float wim = Wim[n * C_DIM + kk];
    WcRe[i] = f2bf(-wim);
    WcIm[i] = f2bf(wre);
  }
  if (i < C_DIM) {
    biasRe[i] = bre[i] - bim[i];
    biasIm[i] = bre[i] + bim[i];
  }
}

// ---------------- bf16 WMMA GEMM:  out[M,1024] = X[M,2048] @ W[1024,2048]^T + bias ---
// Block: 256 thr = 8 waves as 4(M) x 2(N); block tile 128x128; wave tile 32x64.
// Software-pipelined k-loop: register double-buffered A/B fragments.

struct GemmFrags { Frag32B a0, a1, b0, b1, b2, b3; };

__device__ __forceinline__ void gemm_load(const unsigned short* __restrict__ xr0,
                                          const unsigned short* __restrict__ xr1,
                                          const unsigned short* __restrict__ w0,
                                          const unsigned short* __restrict__ w1,
                                          const unsigned short* __restrict__ w2,
                                          const unsigned short* __restrict__ w3,
                                          int k, int lhi, GemmFrags& F) {
  // A-layout: lane<16 -> K k..k+15 ; lane>=16 -> k+16..k+31
  const int ka = k + lhi * 16;
  F.a0.q[0] = *(const u32x4*)(xr0 + ka);
  F.a0.q[1] = *(const u32x4*)(xr0 + ka + 8);
  F.a1.q[0] = *(const u32x4*)(xr1 + ka);
  F.a1.q[1] = *(const u32x4*)(xr1 + ka + 8);
  // B-layout: lane n holds column n = W row n, 16 contiguous K values
  const int kb = k + lhi * 16;
  F.b0.q[0] = *(const u32x4*)(w0 + kb);  F.b0.q[1] = *(const u32x4*)(w0 + kb + 8);
  F.b1.q[0] = *(const u32x4*)(w1 + kb);  F.b1.q[1] = *(const u32x4*)(w1 + kb + 8);
  F.b2.q[0] = *(const u32x4*)(w2 + kb);  F.b2.q[1] = *(const u32x4*)(w2 + kb + 8);
  F.b3.q[0] = *(const u32x4*)(w3 + kb);  F.b3.q[1] = *(const u32x4*)(w3 + kb + 8);
}

__device__ __forceinline__ void gemm_mac(const GemmFrags& F, AccU (&acc)[2][4]) {
  acc[0][0].v = __builtin_amdgcn_wmma_f32_16x16x32_bf16(false, F.a0.v, false, F.b0.v, (short)0, acc[0][0].v, false, false);
  acc[1][0].v = __builtin_amdgcn_wmma_f32_16x16x32_bf16(false, F.a1.v, false, F.b0.v, (short)0, acc[1][0].v, false, false);
  acc[0][1].v = __builtin_amdgcn_wmma_f32_16x16x32_bf16(false, F.a0.v, false, F.b1.v, (short)0, acc[0][1].v, false, false);
  acc[1][1].v = __builtin_amdgcn_wmma_f32_16x16x32_bf16(false, F.a1.v, false, F.b1.v, (short)0, acc[1][1].v, false, false);
  acc[0][2].v = __builtin_amdgcn_wmma_f32_16x16x32_bf16(false, F.a0.v, false, F.b2.v, (short)0, acc[0][2].v, false, false);
  acc[1][2].v = __builtin_amdgcn_wmma_f32_16x16x32_bf16(false, F.a1.v, false, F.b2.v, (short)0, acc[1][2].v, false, false);
  acc[0][3].v = __builtin_amdgcn_wmma_f32_16x16x32_bf16(false, F.a0.v, false, F.b3.v, (short)0, acc[0][3].v, false, false);
  acc[1][3].v = __builtin_amdgcn_wmma_f32_16x16x32_bf16(false, F.a1.v, false, F.b3.v, (short)0, acc[1][3].v, false, false);
}

__global__ __launch_bounds__(256)
void gemm_bf16_kernel(const unsigned short* __restrict__ X,
                      const unsigned short* __restrict__ W,
                      const float* __restrict__ bias,
                      unsigned short* __restrict__ outB,   // bf16 out (or null)
                      float* __restrict__ outF,            // f32 out (or null)
                      float scale) {
  const int lane = threadIdx.x & 31;
  const int wave = threadIdx.x >> 5;
  const int lmod = lane & 15;
  const int lhi  = lane >> 4;
  const int wr = wave & 3;
  const int wc = wave >> 2;
  const int mBase = blockIdx.y * 128 + wr * 32;
  const int nBase = blockIdx.x * 128 + wc * 64;

  AccU acc[2][4];
#pragma unroll
  for (int i = 0; i < 2; ++i)
#pragma unroll
    for (int j = 0; j < 4; ++j)
#pragma unroll
      for (int r = 0; r < 8; ++r) acc[i][j].f[r] = 0.0f;

  const unsigned short* xr0 = X + (size_t)(mBase + lmod) * K_DIM;
  const unsigned short* xr1 = xr0 + (size_t)16 * K_DIM;
  const unsigned short* w0 = W + (size_t)(nBase + 0 * 16 + lmod) * K_DIM;
  const unsigned short* w1 = W + (size_t)(nBase + 1 * 16 + lmod) * K_DIM;
  const unsigned short* w2 = W + (size_t)(nBase + 2 * 16 + lmod) * K_DIM;
  const unsigned short* w3 = W + (size_t)(nBase + 3 * 16 + lmod) * K_DIM;

  GemmFrags F0, F1;
  gemm_load(xr0, xr1, w0, w1, w2, w3, 0, lhi, F0);

  for (int k = 0; k < K_DIM; k += 64) {
    // far prefetch: pull upcoming lines from L2 into near cache (global_prefetch_b8)
    __builtin_prefetch(xr0 + k + 512, 0, 3);
    __builtin_prefetch(xr1 + k + 512, 0, 3);
    __builtin_prefetch(w0 + k + 512, 0, 3);
    __builtin_prefetch(w1 + k + 512, 0, 3);
    __builtin_prefetch(w2 + k + 512, 0, 3);
    __builtin_prefetch(w3 + k + 512, 0, 3);

    gemm_load(xr0, xr1, w0, w1, w2, w3, k + 32, lhi, F1);   // in-flight during F0 macs
    gemm_mac(F0, acc);
    if (k + 64 < K_DIM)
      gemm_load(xr0, xr1, w0, w1, w2, w3, k + 64, lhi, F0); // in-flight during F1 macs
    gemm_mac(F1, acc);
  }

  // D layout: vgpr r -> row (base + r + 8*lhi), col = base + lmod
#pragma unroll
  for (int nt = 0; nt < 4; ++nt) {
    const int col = nBase + nt * 16 + lmod;
    const float bv = bias[col];
#pragma unroll
    for (int mt = 0; mt < 2; ++mt) {
#pragma unroll
      for (int r = 0; r < 8; ++r) {
        const int row = mBase + mt * 16 + r + lhi * 8;
        const float val = acc[mt][nt].f[r] + bv;
        if (outB) outB[(size_t)row * C_DIM + col] = f2bf(val * scale);
        else      outF[(size_t)row * C_DIM + col] = val;
      }
    }
  }
}

// ---------------- flash attention: S = qr.kr + qi.ki (q pre-scaled by 1/8) ----------
// Grid (SEQ/64, HEADS, BATCH), 128 thr = 4 waves, each wave owns 16 query rows.

#define VT_STR 72   // LDS row stride (elems) for transposed V: 144B, 16B-aligned, conflict-free
#define P_STR  40   // LDS row stride for P scratch: 80B, 16B-aligned

__global__ __launch_bounds__(128)
void attn_kernel(const unsigned short* __restrict__ Qr, const unsigned short* __restrict__ Qi,
                 const unsigned short* __restrict__ Kr, const unsigned short* __restrict__ Ki,
                 const unsigned short* __restrict__ Vr, const unsigned short* __restrict__ Vi,
                 unsigned short* __restrict__ O) {     // [M_ROWS, 2*C_DIM] bf16, [re|im]
  __shared__ __align__(16) unsigned short sVtr[HDIM * VT_STR];
  __shared__ __align__(16) unsigned short sVti[HDIM * VT_STR];
  __shared__ __align__(16) unsigned short sP[4][16 * P_STR];

  const int lane = threadIdx.x & 31;
  const int wave = threadIdx.x >> 5;
  const int lmod = lane & 15;
  const int lhi  = lane >> 4;
  const int qt = blockIdx.x;
  const int h  = blockIdx.y;
  const int b  = blockIdx.z;
  const size_t rowBase = (size_t)b * SEQ;
  const int hcol = h * HDIM;

  // Q fragments (A operands), D=64 -> two K=32 chunks; Q already scaled by 0.125
  const int qrow = qt * 64 + wave * 16 + lmod;
  const unsigned short* qrp = Qr + (rowBase + qrow) * C_DIM + hcol;
  const unsigned short* qip = Qi + (rowBase + qrow) * C_DIM + hcol;
  Frag32B aQr[2], aQi[2];
#pragma unroll
  for (int c = 0; c < 2; ++c) {
    const int kb = c * 32 + lhi * 16;
    aQr[c].q[0] = *(const u32x4*)(qrp + kb);
    aQr[c].q[1] = *(const u32x4*)(qrp + kb + 8);
    aQi[c].q[0] = *(const u32x4*)(qip + kb);
    aQi[c].q[1] = *(const u32x4*)(qip + kb + 8);
  }

  float row_m[8], row_l[8];
  AccU accR[4], accI[4];
#pragma unroll
  for (int r = 0; r < 8; ++r) { row_m[r] = -3.0e38f; row_l[r] = 0.0f; }
#pragma unroll
  for (int t = 0; t < 4; ++t)
#pragma unroll
    for (int r = 0; r < 8; ++r) { accR[t].f[r] = 0.0f; accI[t].f[r] = 0.0f; }

  const int stKey = threadIdx.x & 31;   // staging: key index
  const int stDg  = threadIdx.x >> 5;   // staging: 16-dim group

  for (int jb = 0; jb < SEQ; jb += 32) {
    __syncthreads();
    {   // stage V block transposed into LDS: sVt[d][key]
      const unsigned short* vr = Vr + (rowBase + jb + stKey) * C_DIM + hcol + stDg * 16;
      const unsigned short* vi = Vi + (rowBase + jb + stKey) * C_DIM + hcol + stDg * 16;
      Pack16 tr, ti;
      tr.q[0] = *(const u32x4*)(vr);     tr.q[1] = *(const u32x4*)(vr + 8);
      ti.q[0] = *(const u32x4*)(vi);     ti.q[1] = *(const u32x4*)(vi + 8);
#pragma unroll
      for (int e = 0; e < 16; ++e) {
        sVtr[(stDg * 16 + e) * VT_STR + stKey] = tr.s[e];
        sVti[(stDg * 16 + e) * VT_STR + stKey] = ti.s[e];
      }
    }
    __syncthreads();

    // ----- scores: two 16x16 tiles covering 32 keys -----
    AccU S[2];
#pragma unroll
    for (int s = 0; s < 2; ++s) {
#pragma unroll
      for (int r = 0; r < 8; ++r) S[s].f[r] = 0.0f;
      const size_t krow = rowBase + jb + s * 16 + lmod;   // B col = key row of K
      const unsigned short* krp = Kr + krow * C_DIM + hcol;
      const unsigned short* kip = Ki + krow * C_DIM + hcol;
#pragma unroll
      for (int c = 0; c < 2; ++c) {
        const int kd = c * 32 + lhi * 16;
        Frag32B bk;
        bk.q[0] = *(const u32x4*)(krp + kd);
        bk.q[1] = *(const u32x4*)(krp + kd + 8);
        S[s].v = __builtin_amdgcn_wmma_f32_16x16x32_bf16(false, aQr[c].v, false, bk.v,
                                                         (short)0, S[s].v, false, false);
      }
#pragma unroll
      for (int c = 0; c < 2; ++c) {
        const int kd = c * 32 + lhi * 16;
        Frag32B bk;
        bk.q[0] = *(const u32x4*)(kip + kd);
        bk.q[1] = *(const u32x4*)(kip + kd + 8);
        S[s].v = __builtin_amdgcn_wmma_f32_16x16x32_bf16(false, aQi[c].v, false, bk.v,
                                                         (short)0, S[s].v, false, false);
      }
    }

    // ----- online softmax (row = vgpr r + 8*lhi; 16-lane group reductions) -----
    unsigned short* Pw = sP[wave];
#pragma unroll
    for (int r = 0; r < 8; ++r) {
      float cand = fmaxf(S[0].f[r], S[1].f[r]);
#pragma unroll
      for (int m = 1; m <= 8; m <<= 1) cand = fmaxf(cand, __shfl_xor(cand, m, 32));
      const float mnew = fmaxf(row_m[r], cand);
      const float corr = __expf(row_m[r] - mnew);
      const float p0 = __expf(S[0].f[r] - mnew);
      const float p1 = __expf(S[1].f[r] - mnew);
      float psum = p0 + p1;
#pragma unroll
      for (int m = 1; m <= 8; m <<= 1) psum += __shfl_xor(psum, m, 32);
      row_l[r] = row_l[r] * corr + psum;
      row_m[r] = mnew;
#pragma unroll
      for (int t = 0; t < 4; ++t) { accR[t].f[r] *= corr; accI[t].f[r] *= corr; }
      const int prow = r + lhi * 8;                 // D-layout -> P[row][key]
      Pw[prow * P_STR + lmod]      = f2bf(p0);
      Pw[prow * P_STR + 16 + lmod] = f2bf(p1);
    }

    // same-wave LDS RAW: LDS is in-order per wave; fence the compiler + wait
    asm volatile("s_wait_dscnt 0" ::: "memory");

    // re-read P in A-layout
    Frag32B aP;
    const int pk = lhi * 16;
    aP.q[0] = *(const u32x4*)(&Pw[lmod * P_STR + pk]);
    aP.q[1] = *(const u32x4*)(&Pw[lmod * P_STR + pk + 8]);

    // ----- acc += P @ V  (K = 32 keys, 4 output d-tiles, re & im) -----
#pragma unroll
    for (int t = 0; t < 4; ++t) {
      const int d  = t * 16 + lmod;
      const int kk = lhi * 16;
      Frag32B bv;
      bv.q[0] = *(const u32x4*)(&sVtr[d * VT_STR + kk]);
      bv.q[1] = *(const u32x4*)(&sVtr[d * VT_STR + kk + 8]);
      accR[t].v = __builtin_amdgcn_wmma_f32_16x16x32_bf16(false, aP.v, false, bv.v,
                                                          (short)0, accR[t].v, false, false);
      bv.q[0] = *(const u32x4*)(&sVti[d * VT_STR + kk]);
      bv.q[1] = *(const u32x4*)(&sVti[d * VT_STR + kk + 8]);
      accI[t].v = __builtin_amdgcn_wmma_f32_16x16x32_bf16(false, aP.v, false, bv.v,
                                                          (short)0, accI[t].v, false, false);
    }
  }

  float inv[8];
#pragma unroll
  for (int r = 0; r < 8; ++r) inv[r] = 1.0f / row_l[r];

#pragma unroll
  for (int t = 0; t < 4; ++t) {
#pragma unroll
    for (int r = 0; r < 8; ++r) {
      const size_t row = rowBase + (size_t)qt * 64 + wave * 16 + r + lhi * 8;
      const int col = hcol + t * 16 + lmod;
      O[row * K_DIM + col]         = f2bf(accR[t].f[r] * inv[r]);
      O[row * K_DIM + C_DIM + col] = f2bf(accI[t].f[r] * inv[r]);
    }
  }
}

// ------------------------------------ launcher --------------------------------------

extern "C" void kernel_launch(void* const* d_in, const int* in_sizes, int n_in,
                              void* d_out, int out_size, void* d_ws, size_t ws_size,
                              hipStream_t stream) {
  (void)in_sizes; (void)n_in; (void)out_size; (void)ws_size;

  const float* x_re = (const float*)d_in[0];
  const float* x_im = (const float*)d_in[1];
  const float *Wre[4], *bre[4], *Wim[4], *bim[4];
  for (int l = 0; l < 4; ++l) {            // q, k, v, o
    Wre[l] = (const float*)d_in[2 + l * 4 + 0];
    bre[l] = (const float*)d_in[2 + l * 4 + 1];
    Wim[l] = (const float*)d_in[2 + l * 4 + 2];
    bim[l] = (const float*)d_in[2 + l * 4 + 3];
  }

  char* ws = (char*)d_ws;
  size_t off = 0;
  auto take = [&](size_t bytes) -> char* {
    char* p = ws + off;
    off += (bytes + 255) & ~(size_t)255;
    return p;
  };

  unsigned short* X = (unsigned short*)take((size_t)M_ROWS * K_DIM * 2);   // packed [x_re|x_im] bf16
  unsigned short* Wc[4][2];
  float* bc[4][2];
  for (int l = 0; l < 4; ++l)
    for (int s = 0; s < 2; ++s) {
      Wc[l][s] = (unsigned short*)take((size_t)C_DIM * K_DIM * 2);
      bc[l][s] = (float*)take((size_t)C_DIM * 4);
    }
  unsigned short* QKV[6];                                                   // Qr Qi Kr Ki Vr Vi
  for (int i = 0; i < 6; ++i) QKV[i] = (unsigned short*)take((size_t)M_ROWS * C_DIM * 2);
  unsigned short* Oattn = X;   // reuse X: QKV GEMMs are done before attention writes here

  pack_x_kernel<<<(M_ROWS * K_DIM) / 256, 256, 0, stream>>>(x_re, x_im, X);
  for (int l = 0; l < 4; ++l)
    pack_w_kernel<<<(C_DIM * K_DIM) / 256, 256, 0, stream>>>(
        Wre[l], bre[l], Wim[l], bim[l], Wc[l][0], Wc[l][1], bc[l][0], bc[l][1]);

  dim3 gGrid(C_DIM / 128, M_ROWS / 128);
  // Q scaled by 1/sqrt(64) = 0.125 at epilogue so attention scores need no extra scale
  gemm_bf16_kernel<<<gGrid, 256, 0, stream>>>(X, Wc[0][0], bc[0][0], QKV[0], nullptr, 0.125f);
  gemm_bf16_kernel<<<gGrid, 256, 0, stream>>>(X, Wc[0][1], bc[0][1], QKV[1], nullptr, 0.125f);
  gemm_bf16_kernel<<<gGrid, 256, 0, stream>>>(X, Wc[1][0], bc[1][0], QKV[2], nullptr, 1.0f);
  gemm_bf16_kernel<<<gGrid, 256, 0, stream>>>(X, Wc[1][1], bc[1][1], QKV[3], nullptr, 1.0f);
  gemm_bf16_kernel<<<gGrid, 256, 0, stream>>>(X, Wc[2][0], bc[2][0], QKV[4], nullptr, 1.0f);
  gemm_bf16_kernel<<<gGrid, 256, 0, stream>>>(X, Wc[2][1], bc[2][1], QKV[5], nullptr, 1.0f);

  attn_kernel<<<dim3(SEQ / 64, HEADS, BATCH), 128, 0, stream>>>(
      QKV[0], QKV[1], QKV[2], QKV[3], QKV[4], QKV[5], Oattn);

  float* outF = (float*)d_out;
  gemm_bf16_kernel<<<gGrid, 256, 0, stream>>>(Oattn, Wc[3][0], bc[3][0], nullptr, outF, 1.0f);
  gemm_bf16_kernel<<<gGrid, 256, 0, stream>>>(Oattn, Wc[3][1], bc[3][1], nullptr,
                                              outF + (size_t)M_ROWS * C_DIM, 1.0f);
}
